// DeepFMBase_7868380086366
// MI455X (gfx1250) — matrix operand order repaired and checked by reading, not compiled, not run on hardware
//
#include <hip/hip_runtime.h>
#include <hip/hip_bf16.h>

// ---------------------------------------------------------------------------
// DeepFM forward for MI455X (gfx1250, wave32, WMMA).
// Sizes: B=16384, ND=13, NS=26, F=39, K=16, VOCAB=100000, H=400.
// GEMM K-dims padded for v_wmma_f32_16x16x32_f16: feat2 624->640, h0 400->448.
// GEMM: 64x32 wave tiles; K-loop fully unrolled at compile time (template
// KPAD) so the backend software-pipelines loads across WMMAs with no
// accumulator-rotation movs and no spills (__launch_bounds__(128,1)).
// ---------------------------------------------------------------------------

#define Bsz   16384
#define NDf   13
#define NSf   26
#define Ff    39
#define Kdim  16
#define VOC   100000
#define Hdim  400
#define F2    624      // F*K
#define F2P   640      // padded (multiple of 32)
#define HP    448      // padded H (multiple of 32)

typedef __attribute__((ext_vector_type(16))) _Float16 v16h;
typedef __attribute__((ext_vector_type(8)))  float    v8f;

union FragH {           // one WMMA 16-bit A/B fragment: 8 VGPRs / lane
    v16h h;
    int  i[8];
    int4 q[2];
};

// 16-lane butterfly sum (stays inside each half of a wave32)
__device__ __forceinline__ float reduce16(float v) {
    v += __shfl_xor(v, 1);
    v += __shfl_xor(v, 2);
    v += __shfl_xor(v, 4);
    v += __shfl_xor(v, 8);
    return v;
}

// ---------------------------------------------------------------------------
// Transpose f32 weight W[K][N] -> f16 WT[N][Kpad] (zero pad K..Kpad-1).
// ---------------------------------------------------------------------------
__global__ void transpose_pad_f16(const float* __restrict__ W,
                                  _Float16* __restrict__ WT,
                                  int K, int N, int Kpad) {
    int idx = blockIdx.x * 256 + threadIdx.x;
    if (idx >= N * Kpad) return;
    int n  = idx / Kpad;
    int kk = idx - n * Kpad;
    WT[idx] = (kk < K) ? (_Float16)W[(size_t)kk * N + n] : (_Float16)0.0f;
}

// ---------------------------------------------------------------------------
// Feature stage: 16 lanes per sample (lane = k). Embedding rows are 64 B,
// so each gather is one coalesced 64 B transaction per 16-lane group.
// ---------------------------------------------------------------------------
__global__ void build_features(const int*   __restrict__ X,
                               const float* __restrict__ foW,
                               const float* __restrict__ fob,
                               const float* __restrict__ foE,
                               const float* __restrict__ folW,
                               const float* __restrict__ folb,
                               const float* __restrict__ soW,
                               const float* __restrict__ sob,
                               const float* __restrict__ soE,
                               float*    __restrict__ feat2f,
                               _Float16* __restrict__ feat2h,
                               _Float16* __restrict__ h0h,
                               float*    __restrict__ first_out) {
    int t   = threadIdx.x;
    int k   = t & 15;
    int row = blockIdx.x * 16 + (t >> 4);

    const int* xr = X + (size_t)row * Ff;
    float*    f2  = feat2f + (size_t)row * F2;
    _Float16* f2h = feat2h + (size_t)row * F2P;

    float facc = 0.0f;

    // dense fields
    for (int d = 0; d < NDf; ++d) {
        float x   = (float)xr[d];
        float vfo = fmaf(x, foW[d * Kdim + k], fob[d * Kdim + k]);
        facc += reduce16(vfo) * folW[d];
        float vso = fmaf(x, soW[d * Kdim + k], sob[d * Kdim + k]);
        int c = d * Kdim + k;
        f2[c]  = vso;
        f2h[c] = (_Float16)vso;
    }

    // sparse fields (with max-norm)
    for (int s = 0; s < NSf; ++s) {
        int    idx  = xr[NDf + s];
        size_t base = ((size_t)s * VOC + (size_t)idx) * Kdim;

        float e  = foE[base + k];
        float n  = sqrtf(reduce16(e * e));
        float sc = (n > 0.1f) ? 0.1f / (n + 1e-7f) : 1.0f;
        float v  = e * sc;
        facc += reduce16(v) * folW[NDf + s];

        float e2  = soE[base + k];
        float n2  = sqrtf(reduce16(e2 * e2));
        float sc2 = (n2 > 0.1f) ? 0.1f / (n2 + 1e-7f) : 1.0f;
        float v2  = e2 * sc2;
        int c = (NDf + s) * Kdim + k;
        f2[c]  = v2;
        f2h[c] = (_Float16)v2;
    }

    // zero K-padding of feat2h (cols 624..639) and h0h (cols 400..447)
    f2h[F2 + k] = (_Float16)0.0f;
#pragma unroll
    for (int j = 0; j < 3; ++j)
        h0h[(size_t)row * HP + Hdim + j * 16 + k] = (_Float16)0.0f;

    if (k == 0) first_out[row] = facc + folb[0];
}

// ---------------------------------------------------------------------------
// Stage A of the deterministic batch reduction for the FM scalar.
// ---------------------------------------------------------------------------
__global__ void col_partial(const float* __restrict__ feat2f,
                            float* __restrict__ colS,
                            float* __restrict__ colQ) {
    int b  = blockIdx.x;          // 128 blocks x 128 rows
    int c0 = threadIdx.x, c1 = threadIdx.x + 256, c2 = threadIdx.x + 512;
    float s0 = 0, s1 = 0, s2 = 0, q0 = 0, q1 = 0, q2 = 0;
    const float* p = feat2f + (size_t)b * 128 * F2;
    for (int r = 0; r < 128; ++r, p += F2) {
        float v0 = p[c0]; s0 += v0; q0 += v0 * v0;
        float v1 = p[c1]; s1 += v1; q1 += v1 * v1;
        if (c2 < F2) { float v2 = p[c2]; s2 += v2; q2 += v2 * v2; }
    }
    size_t o = (size_t)b * F2;
    colS[o + c0] = s0; colQ[o + c0] = q0;
    colS[o + c1] = s1; colQ[o + c1] = q1;
    if (c2 < F2) { colS[o + c2] = s2; colQ[o + c2] = q2; }
}

// Stage B: second[0] = 0.5 * (sum_c colsum_c^2 - sum_all feat2^2)
__global__ void second_order(const float* __restrict__ colS,
                             const float* __restrict__ colQ,
                             float* __restrict__ second) {
    float s1 = 0, s2 = 0;
    for (int c = threadIdx.x; c < F2; c += 256) {
        float cs = 0, cq = 0;
        for (int b = 0; b < 128; ++b) {
            cs += colS[(size_t)b * F2 + c];
            cq += colQ[(size_t)b * F2 + c];
        }
        s1 += cs * cs;
        s2 += cq;
    }
    __shared__ float sa[256], sb[256];
    sa[threadIdx.x] = s1; sb[threadIdx.x] = s2;
    __syncthreads();
    for (int o = 128; o > 0; o >>= 1) {
        if (threadIdx.x < o) { sa[threadIdx.x] += sa[threadIdx.x + o];
                               sb[threadIdx.x] += sb[threadIdx.x + o]; }
        __syncthreads();
    }
    if (threadIdx.x == 0) second[0] = 0.5f * (sa[0] - sb[0]);
}

// ---------------------------------------------------------------------------
// WMMA GEMM: C = relu(A[M x KPAD] * B + bias), B given as BT[N x KPAD].
// Block = 128 threads (4 waves); each wave owns a 64x32 output tile
// (4 M-subtiles x 2 N-subtiles = 8 named v8f accumulators).
// KPAD is a template constant: the K loop is fully unrolled, strides are
// immediates, and the backend scheduler pipelines each step's 12
// global_load_b128 over the previous step's 8 v_wmma.
//
// A fragment (16x32 f16): row = lane&15, half h = lane>>4,
//   VGPR v holds K = 16*(v>>2) + 8*h + 2*(v&3)  -> two b128 loads.
// B fragment (32x16 f16): col = lane&15, VGPR v holds K = 16*(v>>2) + 8*h + 2*(v&3)
//   -> same K order as A -> two b128 loads.
// D (f32): VGPR v -> row v + 8*h, col = lane&15.
// ---------------------------------------------------------------------------
template <int KPAD>
__global__ __launch_bounds__(128, 1)
void gemm_bias_relu_f16(const _Float16* __restrict__ A,
                        const _Float16* __restrict__ BT,
                        const float*    __restrict__ bias,
                        _Float16*       __restrict__ C,  int ldc,
                        int N) {
    int lane = threadIdx.x & 31;
    int wave = threadIdx.x >> 5;                    // 0..3
    int l16  = lane & 15;
    int h    = lane >> 4;
    int rowBase = blockIdx.x * 256 + wave * 64;
    int colBase = blockIdx.y * 32;
    bool hasN2  = (colBase + 16) < N;               // second 16-col subtile?

    const _Float16* bptr0 = BT + (size_t)(colBase + l16) * KPAD + (h << 3);
    const _Float16* bptr1 = hasN2 ? bptr0 + (size_t)16 * KPAD : bptr0;
    const _Float16* ap0 = A + (size_t)(rowBase      + l16) * KPAD + (h << 3);
    const _Float16* ap1 = A + (size_t)(rowBase + 16 + l16) * KPAD + (h << 3);
    const _Float16* ap2 = A + (size_t)(rowBase + 32 + l16) * KPAD + (h << 3);
    const _Float16* ap3 = A + (size_t)(rowBase + 48 + l16) * KPAD + (h << 3);

    v8f acc0 = (v8f)(0.0f), acc1 = (v8f)(0.0f),
        acc2 = (v8f)(0.0f), acc3 = (v8f)(0.0f),
        acc4 = (v8f)(0.0f), acc5 = (v8f)(0.0f),
        acc6 = (v8f)(0.0f), acc7 = (v8f)(0.0f);

#pragma unroll
    for (int k0 = 0; k0 < KPAD; k0 += 32) {
        FragH b0, b1, a0, a1, a2, a3;
        b0.q[0] = *(const int4*)(bptr0 + k0);
        b0.q[1] = *(const int4*)(bptr0 + k0 + 16);
        b1.q[0] = *(const int4*)(bptr1 + k0);
        b1.q[1] = *(const int4*)(bptr1 + k0 + 16);
        a0.q[0] = *(const int4*)(ap0 + k0);
        a0.q[1] = *(const int4*)(ap0 + k0 + 16);
        a1.q[0] = *(const int4*)(ap1 + k0);
        a1.q[1] = *(const int4*)(ap1 + k0 + 16);
        a2.q[0] = *(const int4*)(ap2 + k0);
        a2.q[1] = *(const int4*)(ap2 + k0 + 16);
        a3.q[0] = *(const int4*)(ap3 + k0);
        a3.q[1] = *(const int4*)(ap3 + k0 + 16);

        acc0 = __builtin_amdgcn_wmma_f32_16x16x32_f16(
            false, a0.h, false, b0.h, (short)0, acc0, false, false);
        acc1 = __builtin_amdgcn_wmma_f32_16x16x32_f16(
            false, a1.h, false, b0.h, (short)0, acc1, false, false);
        acc2 = __builtin_amdgcn_wmma_f32_16x16x32_f16(
            false, a2.h, false, b0.h, (short)0, acc2, false, false);
        acc3 = __builtin_amdgcn_wmma_f32_16x16x32_f16(
            false, a3.h, false, b0.h, (short)0, acc3, false, false);
        acc4 = __builtin_amdgcn_wmma_f32_16x16x32_f16(
            false, a0.h, false, b1.h, (short)0, acc4, false, false);
        acc5 = __builtin_amdgcn_wmma_f32_16x16x32_f16(
            false, a1.h, false, b1.h, (short)0, acc5, false, false);
        acc6 = __builtin_amdgcn_wmma_f32_16x16x32_f16(
            false, a2.h, false, b1.h, (short)0, acc6, false, false);
        acc7 = __builtin_amdgcn_wmma_f32_16x16x32_f16(
            false, a3.h, false, b1.h, (short)0, acc7, false, false);
    }

    // epilogue: bias + relu, f16 store
    {
        int   n  = colBase + l16;
        float bn = bias[n];
        const v8f* accs[4] = { &acc0, &acc1, &acc2, &acc3 };
#pragma unroll
        for (int mt = 0; mt < 4; ++mt) {
#pragma unroll
            for (int v = 0; v < 8; ++v) {
                int m = rowBase + mt * 16 + (h << 3) + v;
                float val = fmaxf((*accs[mt])[v] + bn, 0.0f);
                C[(size_t)m * ldc + n] = (_Float16)val;
            }
        }
    }
    if (hasN2) {
        int   n  = colBase + 16 + l16;
        float bn = bias[n];
        const v8f* accs[4] = { &acc4, &acc5, &acc6, &acc7 };
#pragma unroll
        for (int mt = 0; mt < 4; ++mt) {
#pragma unroll
            for (int v = 0; v < 8; ++v) {
                int m = rowBase + mt * 16 + (h << 3) + v;
                float val = fmaxf((*accs[mt])[v] + bn, 0.0f);
                C[(size_t)m * ldc + n] = (_Float16)val;
            }
        }
    }
}

// ---------------------------------------------------------------------------
// out[i] = first_out[i] + second + dot(relu_h1[i,:400], W2) + b2
// ---------------------------------------------------------------------------
__global__ void final_out(const _Float16* __restrict__ h1h,
                          const float* __restrict__ W2,
                          const float* __restrict__ b2,
                          const float* __restrict__ first_out,
                          const float* __restrict__ second,
                          float* __restrict__ out) {
    int lane = threadIdx.x & 31;
    int row  = blockIdx.x * 8 + (threadIdx.x >> 5);
    const _Float16* hr = h1h + (size_t)row * Hdim;
    float acc = 0.0f;
    for (int kk = lane; kk < Hdim; kk += 32)
        acc += (float)hr[kk] * W2[kk];
    acc += __shfl_xor(acc, 1);
    acc += __shfl_xor(acc, 2);
    acc += __shfl_xor(acc, 4);
    acc += __shfl_xor(acc, 8);
    acc += __shfl_xor(acc, 16);
    if (lane == 0)
        out[row] = acc + first_out[row] + second[0] + b2[0];
}

// ---------------------------------------------------------------------------
extern "C" void kernel_launch(void* const* d_in, const int* in_sizes, int n_in,
                              void* d_out, int out_size, void* d_ws, size_t ws_size,
                              hipStream_t stream) {
    const int*   X    = (const int*)  d_in[0];
    const float* foW  = (const float*)d_in[1];
    const float* fob  = (const float*)d_in[2];
    const float* foE  = (const float*)d_in[3];
    const float* folW = (const float*)d_in[4];
    const float* folb = (const float*)d_in[5];
    const float* soW  = (const float*)d_in[6];
    const float* sob  = (const float*)d_in[7];
    const float* soE  = (const float*)d_in[8];
    const float* W0   = (const float*)d_in[9];
    const float* b0   = (const float*)d_in[10];
    const float* W1   = (const float*)d_in[11];
    const float* b1   = (const float*)d_in[12];
    const float* W2   = (const float*)d_in[13];
    const float* b2   = (const float*)d_in[14];
    float* out = (float*)d_out;

    // carve workspace (256-B aligned slices), ~91 MB total
    char*  ws  = (char*)d_ws;
    size_t off = 0;
    auto carve = [&](size_t bytes) -> void* {
        void* p = ws + off;
        off = (off + bytes + 255) & ~(size_t)255;
        return p;
    };
    float*    feat2f = (float*)   carve((size_t)Bsz * F2  * 4);
    _Float16* feat2h = (_Float16*)carve((size_t)Bsz * F2P * 2);
    _Float16* W0T    = (_Float16*)carve((size_t)Hdim * F2P * 2);
    _Float16* W1T    = (_Float16*)carve((size_t)Hdim * HP  * 2);
    _Float16* h0h    = (_Float16*)carve((size_t)Bsz * HP   * 2);
    _Float16* h1h    = (_Float16*)carve((size_t)Bsz * Hdim * 2);
    float*    first  = (float*)   carve((size_t)Bsz * 4);
    float*    colS   = (float*)   carve((size_t)128 * F2 * 4);
    float*    colQ   = (float*)   carve((size_t)128 * F2 * 4);
    float*    second = (float*)   carve(256);

    // 1) weight transposes (f32 -> padded f16 BT layout)
    transpose_pad_f16<<<(Hdim * F2P + 255) / 256, 256, 0, stream>>>(
        W0, W0T, F2, Hdim, F2P);
    transpose_pad_f16<<<(Hdim * HP + 255) / 256, 256, 0, stream>>>(
        W1, W1T, Hdim, Hdim, HP);

    // 2) features + first order (+ zero pads)
    build_features<<<Bsz / 16, 256, 0, stream>>>(
        X, foW, fob, foE, folW, folb, soW, sob, soE,
        feat2f, feat2h, h0h, first);

    // 3) FM second-order scalar (deterministic two-stage reduction)
    col_partial<<<128, 256, 0, stream>>>(feat2f, colS, colQ);
    second_order<<<1, 256, 0, stream>>>(colS, colQ, second);

    // 4) deep MLP: h0 = relu(feat2 @ W0 + b0) ; h1 = relu(h0 @ W1 + b1)
    dim3 g0(Bsz / 256, (Hdim + 31) / 32);           // (64, 13)
    gemm_bias_relu_f16<F2P><<<g0, 128, 0, stream>>>(feat2h, W0T, b0,
                                                    h0h, HP, Hdim);
    gemm_bias_relu_f16<HP><<<g0, 128, 0, stream>>>(h0h, W1T, b1,
                                                   h1h, Hdim, Hdim);

    // 5) final: dot with W2 + first order + FM scalar + b2
    final_out<<<Bsz / 8, 256, 0, stream>>>(h1h, W2, b2, first, second, out);

    (void)in_sizes; (void)n_in; (void)out_size; (void)ws_size;
}
